// Self_Attention_RoPE_79362405695738
// MI455X (gfx1250) — compile-verified, matrix operand order checked by reading, not run on hardware
//
#include <hip/hip_runtime.h>

typedef __attribute__((ext_vector_type(16))) __bf16        v16bf;
typedef __attribute__((ext_vector_type(8)))  float         v8f;
typedef __attribute__((ext_vector_type(4)))  unsigned int  u32x4;

#define S_LEN   2048
#define DMODEL  2048
#define NHEADS  16
#define DHEAD   128
#define BATCH   4
#define NEG_BIG (-3.0e38f)

__device__ __forceinline__ unsigned short f2bf(float f) {
  unsigned int u = __float_as_uint(f);
  u += 0x7FFFu + ((u >> 16) & 1u);          // round-to-nearest-even
  return (unsigned short)(u >> 16);
}
__device__ __forceinline__ float bf2f(unsigned short h) {
  return __uint_as_float(((unsigned int)h) << 16);
}
__device__ __forceinline__ unsigned int pk2bf(float lo, float hi) {
  return (unsigned int)f2bf(lo) | ((unsigned int)f2bf(hi) << 16);
}

// Build a 16xbf16 WMMA operand fragment from two 16-byte chunks.
// Per-lane layout (ISA 16-bit A/B 16x32): lane&15 = row, (lane>>4)*8 = K-half
// offset; VGPR0..3 <- K = koff+0..7, VGPR4..7 <- K = koff+16..23.
__device__ __forceinline__ v16bf mkfrag(const unsigned short* lo,
                                        const unsigned short* hi) {
  union { u32x4 u[2]; v16bf v; } f;
  f.u[0] = *reinterpret_cast<const u32x4*>(lo);
  f.u[1] = *reinterpret_cast<const u32x4*>(hi);
  return f.v;
}

__device__ __forceinline__ v8f wmma_bf16(v16bf a, v16bf b, v8f c) {
  return __builtin_amdgcn_wmma_f32_16x16x32_bf16(
      /*neg_a=*/false, a, /*neg_b=*/false, b,
      /*c_mod=*/(short)0, c, /*reuse_a=*/false, /*reuse_b=*/false);
}

// ---------------------------------------------------------------- cvt f32->bf16
__global__ __launch_bounds__(256) void cvt_kernel(const float* __restrict__ src,
                                                  unsigned short* __restrict__ dst,
                                                  int n) {
  int i = blockIdx.x * 256 + threadIdx.x;
  if (i < n) dst[i] = f2bf(src[i]);
}

// ------------------------------------------------------------- QKV projections
// One wave computes a 32(M) x 64(N) tile.  mode = blockIdx.z: 0=Q 1=K 2=V.
// y[m,n] = sum_d x[m,d] * w[n,d]   (einsum 'bsd,ed->bse')
__global__ __launch_bounds__(32) void gemm_qkv_kernel(
    const unsigned short* __restrict__ Xb,
    const unsigned short* __restrict__ W0,   // Wq, with Wk/Wv contiguous after
    unsigned short* __restrict__ Qb,
    unsigned short* __restrict__ Kb,
    unsigned short* __restrict__ Vt) {
  const int lane  = threadIdx.x & 31;
  const int mode  = blockIdx.z;
  const int lr    = lane & 15;
  const int koff  = (lane >> 4) * 8;
  const int radd  = (lane >> 4) * 8;
  const int mbase = blockIdx.x * 32;
  const int nbase = blockIdx.y * 64;
  const unsigned short* Wm = W0 + (size_t)mode * DMODEL * DMODEL;

  const unsigned short* arow[2];
  const unsigned short* brow[4];
#pragma unroll
  for (int mt = 0; mt < 2; ++mt)
    arow[mt] = Xb + (size_t)(mbase + 16 * mt + lr) * DMODEL;
#pragma unroll
  for (int nt = 0; nt < 4; ++nt)
    brow[nt] = Wm + (size_t)(nbase + 16 * nt + lr) * DMODEL;

  v8f acc[2][4];
#pragma unroll
  for (int mt = 0; mt < 2; ++mt)
#pragma unroll
    for (int nt = 0; nt < 4; ++nt)
      acc[mt][nt] = (v8f){0.f, 0.f, 0.f, 0.f, 0.f, 0.f, 0.f, 0.f};

  for (int kc = 0; kc < DMODEL; kc += 32) {
    v16bf av[2], bv[4];
#pragma unroll
    for (int mt = 0; mt < 2; ++mt)
      av[mt] = mkfrag(arow[mt] + kc + koff, arow[mt] + kc + koff + 16);
#pragma unroll
    for (int nt = 0; nt < 4; ++nt)
      bv[nt] = mkfrag(brow[nt] + kc + koff, brow[nt] + kc + koff + 16);
#pragma unroll
    for (int mt = 0; mt < 2; ++mt)
#pragma unroll
      for (int nt = 0; nt < 4; ++nt)
        acc[mt][nt] = wmma_bf16(av[mt], bv[nt], acc[mt][nt]);
  }

  // C layout: VGPR r, lanes 0-15 -> M=r ; lanes 16-31 -> M=r+8 ; N = lane&15.
  // Uniform branch on mode (hoisted out of the store loops).
  if (mode < 2) {
    unsigned short* dst = (mode == 0) ? Qb : Kb;
#pragma unroll
    for (int mt = 0; mt < 2; ++mt) {
#pragma unroll
      for (int nt = 0; nt < 4; ++nt) {
#pragma unroll
        for (int r = 0; r < 8; ++r) {
          const int m = mbase + 16 * mt + r + radd;
          const int n = nbase + 16 * nt + lr;
          const int b = m >> 11, s = m & (S_LEN - 1);
          const int h = n >> 7,  d = n & (DHEAD - 1);
          dst[(((size_t)(b * NHEADS + h)) * S_LEN + s) * DHEAD + d] =
              f2bf(acc[mt][nt][r]);
        }
      }
    }
  } else {
    // V^T: consecutive r <-> consecutive s  => pack 8 bf16 into one b128 store
#pragma unroll
    for (int mt = 0; mt < 2; ++mt) {
#pragma unroll
      for (int nt = 0; nt < 4; ++nt) {
        const int m0 = mbase + 16 * mt + radd;          // row of r=0
        const int n  = nbase + 16 * nt + lr;
        const int b = m0 >> 11, s0 = m0 & (S_LEN - 1);  // s0 multiple of 8
        const int h = n >> 7,   d  = n & (DHEAD - 1);
        union { unsigned int w[4]; u32x4 v; } pk;
#pragma unroll
        for (int j = 0; j < 4; ++j)
          pk.w[j] = pk2bf(acc[mt][nt][2 * j], acc[mt][nt][2 * j + 1]);
        *reinterpret_cast<u32x4*>(
            Vt + (((size_t)(b * NHEADS + h)) * DHEAD + d) * S_LEN + s0) = pk.v;
      }
    }
  }
}

// -------------------------------------------------------------------- RoPE
// In-place pairwise rotation on Q and K ([b,h,s,dh] bf16).  Q also absorbs
// the 1/sqrt(dh) score scale.
__global__ __launch_bounds__(256) void rope_kernel(unsigned short* __restrict__ Qb,
                                                   unsigned short* __restrict__ Kb,
                                                   const int* __restrict__ pos) {
  const int C = BATCH * NHEADS * S_LEN * (DHEAD / 2);  // 8388608
  int id = blockIdx.x * 256 + threadIdx.x;
  if (id >= 2 * C) return;
  unsigned short* base = (id < C) ? Qb : Kb;
  const float scale = (id < C) ? 0.08838834764831845f : 1.0f;  // 1/sqrt(128)
  const int r  = (id < C) ? id : id - C;
  const int i  = r & 63;
  const int s  = (r >> 6) & (S_LEN - 1);
  const int bh = r >> 17;
  const float p = (float)pos[s];
  // theta^(-2i/dh) = exp(-i * ln(1e4)/64)
  const float ang = p * __expf(-(float)i * (9.210340371976184f / 64.0f));
  float sn, cs;
  __sincosf(ang, &sn, &cs);
  const size_t a = (((size_t)bh * S_LEN + s) * DHEAD) + 2 * i;
  const float x1 = bf2f(base[a]);
  const float x2 = bf2f(base[a + 1]);
  base[a]     = f2bf((x1 * cs - x2 * sn) * scale);
  base[a + 1] = f2bf((x1 * sn + x2 * cs) * scale);
}

// ------------------------------------------------------------ flash attention
// 4 waves/block; wave owns one 16-row q-tile of one (b,h).  Online softmax
// over causal k-chunks of 32.  Output bf16 to Ao[b,s,h*dh+d].
__global__ __launch_bounds__(128) void flash_kernel(
    const unsigned short* __restrict__ Qb,
    const unsigned short* __restrict__ Kb,
    const unsigned short* __restrict__ Vt,
    unsigned short* __restrict__ Ao) {
  __shared__ __align__(16) unsigned short pbuf[4][16][32];

  const int lane  = threadIdx.x & 31;
  const int wv    = threadIdx.x >> 5;
  const int lr    = lane & 15;
  const int koff  = (lane >> 4) * 8;
  const int radd  = (lane >> 4) * 8;
  const int bh    = blockIdx.y;
  const int qbase = (blockIdx.x * 4 + wv) * 16;

  const unsigned short* Qp = Qb + (size_t)bh * S_LEN * DHEAD;
  const unsigned short* Kp = Kb + (size_t)bh * S_LEN * DHEAD;
  const unsigned short* Vp = Vt + (size_t)bh * DHEAD * S_LEN;

  // Q A-fragments for the whole tile (dh = 128 -> 4 K-steps of 32), invariant.
  v16bf qa[4];
  {
    const unsigned short* qrow = Qp + (size_t)(qbase + lr) * DHEAD;
#pragma unroll
    for (int dc = 0; dc < 4; ++dc)
      qa[dc] = mkfrag(qrow + dc * 32 + koff, qrow + dc * 32 + koff + 16);
  }

  v8f acc[8];
#pragma unroll
  for (int t = 0; t < 8; ++t)
    acc[t] = (v8f){0.f, 0.f, 0.f, 0.f, 0.f, 0.f, 0.f, 0.f};
  float m8[8], l8[8], a8[8];
#pragma unroll
  for (int r = 0; r < 8; ++r) { m8[r] = NEG_BIG; l8[r] = 0.f; }

  const int kend = qbase + 16;
  for (int kc = 0; kc < kend; kc += 32) {
    // ---- scores: 16q x 32k  (two 16x16 N-tiles)
    v8f s0 = (v8f){0.f, 0.f, 0.f, 0.f, 0.f, 0.f, 0.f, 0.f};
    v8f s1 = (v8f){0.f, 0.f, 0.f, 0.f, 0.f, 0.f, 0.f, 0.f};
    const unsigned short* krow0 = Kp + (size_t)(kc + lr) * DHEAD;
    const unsigned short* krow1 = krow0 + 16 * DHEAD;
#pragma unroll
    for (int dc = 0; dc < 4; ++dc) {
      v16bf kb0 = mkfrag(krow0 + dc * 32 + koff, krow0 + dc * 32 + koff + 16);
      v16bf kb1 = mkfrag(krow1 + dc * 32 + koff, krow1 + dc * 32 + koff + 16);
      s0 = wmma_bf16(qa[dc], kb0, s0);
      s1 = wmma_bf16(qa[dc], kb1, s1);
    }

    // ---- causal mask + online softmax (rows live in 16-lane halves)
#pragma unroll
    for (int r = 0; r < 8; ++r) {
      const int qr = qbase + r + radd;
      const int k0 = kc + lr, k1 = k0 + 16;
      float x0 = (k0 <= qr) ? s0[r] : NEG_BIG;
      float x1 = (k1 <= qr) ? s1[r] : NEG_BIG;
      float v = fmaxf(x0, x1);
      v = fmaxf(v, __shfl_xor(v, 1));
      v = fmaxf(v, __shfl_xor(v, 2));
      v = fmaxf(v, __shfl_xor(v, 4));
      v = fmaxf(v, __shfl_xor(v, 8));
      const float mn = fmaxf(m8[r], v);
      const float al = __expf(m8[r] - mn);
      m8[r] = mn;
      const float p0 = __expf(x0 - mn);
      const float p1 = __expf(x1 - mn);
      s0[r] = p0;
      s1[r] = p1;
      float rs = p0 + p1;
      rs += __shfl_xor(rs, 1);
      rs += __shfl_xor(rs, 2);
      rs += __shfl_xor(rs, 4);
      rs += __shfl_xor(rs, 8);
      l8[r] = l8[r] * al + rs;
      a8[r] = al;
    }
#pragma unroll
    for (int t = 0; t < 8; ++t)
#pragma unroll
      for (int r = 0; r < 8; ++r)
        acc[t][r] *= a8[r];

    // ---- stage P (f32 C-layout) -> LDS bf16 row-major 16x32
#pragma unroll
    for (int r = 0; r < 8; ++r) {
      pbuf[wv][r + radd][lr]      = f2bf(s0[r]);
      pbuf[wv][r + radd][lr + 16] = f2bf(s1[r]);
    }
    asm volatile("s_wait_dscnt 0" ::: "memory");  // lane-crossing LDS RAW
    const v16bf pa = mkfrag(&pbuf[wv][lr][koff], &pbuf[wv][lr][koff + 16]);

    // ---- out += P @ V   (V stored transposed: rows are dh, cols are s)
#pragma unroll
    for (int t = 0; t < 8; ++t) {
      const unsigned short* vrow = Vp + (size_t)(t * 16 + lr) * S_LEN + kc;
      v16bf vb = mkfrag(vrow + koff, vrow + koff + 16);
      acc[t] = wmma_bf16(pa, vb, acc[t]);
    }
  }

  // ---- finalize: divide by l, emit bf16 to Ao[b, s, h*dh + d]
  const int b = bh >> 4, h = bh & 15;
#pragma unroll
  for (int t = 0; t < 8; ++t) {
#pragma unroll
    for (int r = 0; r < 8; ++r) {
      const int srow = qbase + r + radd;
      const int d = t * 16 + lr;
      const float o = acc[t][r] / l8[r];
      Ao[((size_t)(b * S_LEN + srow)) * DMODEL + h * DHEAD + d] = f2bf(o);
    }
  }
}

// ------------------------------------------------------------ output projection
__global__ __launch_bounds__(32) void gemm_out_kernel(
    const unsigned short* __restrict__ Ao,
    const unsigned short* __restrict__ Wo,
    float* __restrict__ out) {
  const int lane  = threadIdx.x & 31;
  const int lr    = lane & 15;
  const int koff  = (lane >> 4) * 8;
  const int radd  = (lane >> 4) * 8;
  const int mbase = blockIdx.x * 32;
  const int nbase = blockIdx.y * 64;

  const unsigned short* arow[2];
  const unsigned short* brow[4];
#pragma unroll
  for (int mt = 0; mt < 2; ++mt)
    arow[mt] = Ao + (size_t)(mbase + 16 * mt + lr) * DMODEL;
#pragma unroll
  for (int nt = 0; nt < 4; ++nt)
    brow[nt] = Wo + (size_t)(nbase + 16 * nt + lr) * DMODEL;

  v8f acc[2][4];
#pragma unroll
  for (int mt = 0; mt < 2; ++mt)
#pragma unroll
    for (int nt = 0; nt < 4; ++nt)
      acc[mt][nt] = (v8f){0.f, 0.f, 0.f, 0.f, 0.f, 0.f, 0.f, 0.f};

  for (int kc = 0; kc < DMODEL; kc += 32) {
    v16bf av[2], bv[4];
#pragma unroll
    for (int mt = 0; mt < 2; ++mt)
      av[mt] = mkfrag(arow[mt] + kc + koff, arow[mt] + kc + koff + 16);
#pragma unroll
    for (int nt = 0; nt < 4; ++nt)
      bv[nt] = mkfrag(brow[nt] + kc + koff, brow[nt] + kc + koff + 16);
#pragma unroll
    for (int mt = 0; mt < 2; ++mt)
#pragma unroll
      for (int nt = 0; nt < 4; ++nt)
        acc[mt][nt] = wmma_bf16(av[mt], bv[nt], acc[mt][nt]);
  }

#pragma unroll
  for (int mt = 0; mt < 2; ++mt)
#pragma unroll
    for (int nt = 0; nt < 4; ++nt)
#pragma unroll
      for (int r = 0; r < 8; ++r) {
        const int m = mbase + 16 * mt + r + radd;
        const int n = nbase + 16 * nt + lr;
        out[(size_t)m * DMODEL + n] = acc[mt][nt][r];
      }
}

// ------------------------------------------------------------------- launch
extern "C" void kernel_launch(void* const* d_in, const int* in_sizes, int n_in,
                              void* d_out, int out_size, void* d_ws, size_t ws_size,
                              hipStream_t stream) {
  const float* x   = (const float*)d_in[0];
  const int*   pos = (const int*)d_in[1];
  const float* wq  = (const float*)d_in[2];
  const float* wk  = (const float*)d_in[3];
  const float* wv  = (const float*)d_in[4];
  const float* wo  = (const float*)d_in[5];
  float*       out = (float*)d_out;

  const size_t XE = (size_t)BATCH * S_LEN * DMODEL;   // 16,777,216
  const size_t WE = (size_t)DMODEL * DMODEL;          //  4,194,304

  unsigned short* Xb = (unsigned short*)d_ws;
  unsigned short* Wq = Xb + XE;
  unsigned short* Wk = Wq + WE;
  unsigned short* Wv = Wk + WE;
  unsigned short* Wo = Wv + WE;
  unsigned short* Qb = Wo + WE;
  unsigned short* Kb = Qb + XE;
  unsigned short* Vt = Kb + XE;
  unsigned short* Ao = Vt + XE;   // total ~201 MB

  // 1) downconvert inputs to bf16
  cvt_kernel<<<(int)((XE + 255) / 256), 256, 0, stream>>>(x,  Xb, (int)XE);
  cvt_kernel<<<(int)((WE + 255) / 256), 256, 0, stream>>>(wq, Wq, (int)WE);
  cvt_kernel<<<(int)((WE + 255) / 256), 256, 0, stream>>>(wk, Wk, (int)WE);
  cvt_kernel<<<(int)((WE + 255) / 256), 256, 0, stream>>>(wv, Wv, (int)WE);
  cvt_kernel<<<(int)((WE + 255) / 256), 256, 0, stream>>>(wo, Wo, (int)WE);

  // 2) QKV projections (z selects weight/dest)
  {
    dim3 grid(BATCH * S_LEN / 32, DMODEL / 64, 3);
    gemm_qkv_kernel<<<grid, 32, 0, stream>>>(Xb, Wq, Qb, Kb, Vt);
  }

  // 3) RoPE (in place; Q absorbs 1/sqrt(dh))
  {
    const int total = 2 * BATCH * NHEADS * S_LEN * (DHEAD / 2);
    rope_kernel<<<(total + 255) / 256, 256, 0, stream>>>(Qb, Kb, pos);
  }

  // 4) causal flash attention
  {
    dim3 grid(S_LEN / 16 / 4, BATCH * NHEADS);
    flash_kernel<<<grid, 128, 0, stream>>>(Qb, Kb, Vt, Ao);
  }

  // 5) output projection -> f32
  {
    dim3 grid(BATCH * S_LEN / 32, DMODEL / 64);
    gemm_out_kernel<<<grid, 32, 0, stream>>>(Ao, Wo, out);
  }
}
